// _AttentionCore_27882927686301
// MI455X (gfx1250) — matrix operand order repaired and checked by reading, not compiled, not run on hardware
//
#include <hip/hip_runtime.h>
#include <math.h>

typedef __attribute__((ext_vector_type(16))) _Float16 v16h;
typedef __attribute__((ext_vector_type(8)))  float    v8f;
typedef __attribute__((ext_vector_type(2)))  float    v2f;

#define ATTN_SCALE 0.125f
#define NEG_BIG   -3.402823466e38f   // jnp.finfo(f32).min

// One wave32 per token. B,H,T,K,D = 2,8,4096,32,64  ->  65536 tokens.
// Scores via V_WMMA_F32_16X16X32_F16 with split-f16 error compensation
// (f32-grade accuracy, f32 accumulator). Value projection + self-value
// rejection in coalesced f32 VALU (memory-bound kernel, 0.5 FLOP/B).
__global__ __launch_bounds__(256)
void attn_excl_kernel(const float* __restrict__ Q,
                      const float* __restrict__ Kv,
                      const float* __restrict__ Vv,
                      const int*   __restrict__ Mv,
                      const float* __restrict__ SV,
                      float*       __restrict__ O,
                      int ntok)
{
    const int lane = threadIdx.x & 31;
    const int wid  = threadIdx.x >> 5;
    const int tok  = blockIdx.x * 8 + wid;
    if (tok >= ntok) return;               // wave-uniform: EXEC stays all-ones

    const int g    = lane >> 4;            // lane half-group (0 or 1)
    const int mrow = lane & 15;            // A-matrix row index within half

    const float* qp = Q  + (size_t)tok * 64;
    const float* kp = Kv + (size_t)tok * 2048;
    const float* vp = Vv + (size_t)tok * 2048;
    const int*   mp = Mv + (size_t)tok * 32;
    const float* sp = SV + (size_t)tok * 64;
    float*       op = O  + (size_t)tok * 64;

    // Warm the value tile (8KB) while score WMMAs run: global_prefetch_b8
    __builtin_prefetch(vp + lane * 64, 0, 0);

    // ---------- B operand: query replicated across all 16 columns ----------
    // B(32x16) f16 layout: lanes 0-15 hold rows c=0..15, lanes 16-31 rows
    // c=16..31; element i of v16h = row (16*g + i). Value = q[dbase + c] for
    // every column -> each D column is identical (no score gather needed).
    v16h Bh[2], Bl[2];
#pragma unroll
    for (int db = 0; db < 2; ++db) {
        const float* qq = qp + db * 32 + g * 16;
#pragma unroll
        for (int i = 0; i < 16; ++i) {
            float     x  = qq[i];
            _Float16  h  = (_Float16)x;
            Bh[db][i]    = h;
            Bl[db][i]    = (_Float16)(x - (float)h);
        }
    }

    // ---------- scores: two key-halves (rows 0-15, 16-31) ----------
    // A(16x32) f16 layout per lane: row = l%16; element i -> d = dbase+16g+i.
    float sc[2][8];
#pragma unroll
    for (int kh = 0; kh < 2; ++kh) {
        v8f C = {0.f, 0.f, 0.f, 0.f, 0.f, 0.f, 0.f, 0.f};
#pragma unroll
        for (int db = 0; db < 2; ++db) {
            const float* kr = kp + (size_t)(kh * 16 + mrow) * 64 + db * 32 + g * 16;
            v16h Ah, Al;
#pragma unroll
            for (int i = 0; i < 8; ++i) {
                float    x0 = kr[i];
                float    x1 = kr[8 + i];
                _Float16 h0 = (_Float16)x0;
                _Float16 h1 = (_Float16)x1;
                Ah[i]     = h0;  Al[i]     = (_Float16)(x0 - (float)h0);
                Ah[8 + i] = h1;  Al[8 + i] = (_Float16)(x1 - (float)h1);
            }
            // split-f16: k*q ~= kh*qh + kl*qh + kh*ql  (f32 accumulate)
            C = __builtin_amdgcn_wmma_f32_16x16x32_f16(false, Ah, false, Bh[db],
                                                       (short)0, C, false, false);
            C = __builtin_amdgcn_wmma_f32_16x16x32_f16(false, Al, false, Bh[db],
                                                       (short)0, C, false, false);
            C = __builtin_amdgcn_wmma_f32_16x16x32_f16(false, Ah, false, Bl[db],
                                                       (short)0, C, false, false);
        }
        // D element r (any column) = score for key kh*16 + 8g + r
#pragma unroll
        for (int r = 0; r < 8; ++r) {
            int   kidx = kh * 16 + g * 8 + r;
            float v    = C[r] * ATTN_SCALE;
            sc[kh][r]  = mp[kidx] ? v : NEG_BIG;
        }
    }

    // ---------- masked softmax over K=32 (each lane holds 16 scores) ----------
    float mx = NEG_BIG;
#pragma unroll
    for (int r = 0; r < 8; ++r) {
        mx = fmaxf(mx, sc[0][r]);
        mx = fmaxf(mx, sc[1][r]);
    }
    mx = fmaxf(mx, __shfl_xor(mx, 16, 32));

    float e[2][8];
    float sum = 0.f;
#pragma unroll
    for (int kh = 0; kh < 2; ++kh)
#pragma unroll
        for (int r = 0; r < 8; ++r) {
            float ev = __expf(sc[kh][r] - mx);
            e[kh][r] = ev;
            sum += ev;
        }
    sum += __shfl_xor(sum, 16, 32);
    const float inv = 1.0f / sum;

    // assemble all 32 attention weights per lane via cross-half exchange
    float a[32];
#pragma unroll
    for (int r = 0; r < 8; ++r) {
        float own1 = e[0][r] * inv;            // key  8g + r
        float own2 = e[1][r] * inv;            // key 16 + 8g + r
        float oth1 = __shfl_xor(own1, 16, 32); // partner half's keys
        float oth2 = __shfl_xor(own2, 16, 32);
        a[r]      = g ? oth1 : own1;
        a[8 + r]  = g ? own1 : oth1;
        a[16 + r] = g ? oth2 : own2;
        a[24 + r] = g ? own2 : oth2;
    }

    // ---------- out[d] = sum_k a[k] * V[k][d]; lane owns d = 2*lane, 2*lane+1 ----------
    float o0 = 0.f, o1 = 0.f;
#pragma unroll
    for (int k = 0; k < 32; ++k) {
        v2f vv = *(const v2f*)(vp + (size_t)k * 64 + 2 * lane);  // 256B/row coalesced
        o0 = fmaf(a[k], vv.x, o0);
        o1 = fmaf(a[k], vv.y, o1);
    }

    // ---------- remove component along normalized self_value ----------
    v2f s2 = *(const v2f*)(sp + 2 * lane);
    float ss = s2.x * s2.x + s2.y * s2.y;
#pragma unroll
    for (int off = 16; off > 0; off >>= 1) ss += __shfl_xor(ss, off, 32);
    float nrm  = sqrtf(ss);
    float dinv = 1.0f / fmaxf(nrm, 1e-12f);
    float u0 = s2.x * dinv, u1 = s2.y * dinv;

    float pd = o0 * u0 + o1 * u1;
#pragma unroll
    for (int off = 16; off > 0; off >>= 1) pd += __shfl_xor(pd, off, 32);

    o0 -= pd * u0;
    o1 -= pd * u1;

    v2f ov; ov.x = o0; ov.y = o1;
    *(v2f*)(op + 2 * lane) = ov;
}

extern "C" void kernel_launch(void* const* d_in, const int* in_sizes, int n_in,
                              void* d_out, int out_size, void* d_ws, size_t ws_size,
                              hipStream_t stream) {
    const float* q    = (const float*)d_in[0];  // (B,H,T,D)
    const float* k    = (const float*)d_in[1];  // (B,H,T,K,D)
    const float* v    = (const float*)d_in[2];  // (B,H,T,K,D)
    const int*   mask = (const int*)  d_in[3];  // (B,H,T,K)
    const float* sv   = (const float*)d_in[4];  // (B,H,T,D)
    float*       out  = (float*)d_out;          // (B,H,T,D)

    const int ntok   = in_sizes[0] / 64;        // B*H*T = 65536
    const int blocks = (ntok + 7) / 8;          // 8 waves (tokens) per block

    attn_excl_kernel<<<blocks, 256, 0, stream>>>(q, k, v, mask, sv, out, ntok);
}
